// GCGRU_38620345925852
// MI455X (gfx1250) — matrix operand order repaired and checked by reading, not compiled, not run on hardware
//
#include <hip/hip_runtime.h>
#include <math.h>

// ---------------- types ----------------
typedef __bf16 bf16;
typedef bf16  v16bf  __attribute__((ext_vector_type(16)));
typedef bf16  bf16x8 __attribute__((ext_vector_type(8)));
typedef float v8f    __attribute__((ext_vector_type(8)));

union FragU { bf16x8 h[2]; v16bf v; };
__device__ __forceinline__ v16bf make_frag(bf16x8 lo, bf16x8 hi) {
  FragU u; u.h[0] = lo; u.h[1] = hi; return u.v;
}

// ---------------- problem constants ----------------
#define NB   64
#define NN   1024
#define NC   128
#define NE   16
#define NKI  384
#define XG_KSTRIDE   ((size_t)NB * NN * NC)   // 8388608 elems per support slice
#define XG_BSTRIDE   ((size_t)NN * NC)        // 131072 elems per batch

#define WMMA_BF16(a, b, c) \
  __builtin_amdgcn_wmma_f32_16x16x32_bf16(false, (a), false, (b), (short)0, (c), false, false)

// =============== 32x64 wave-tile GEMM core (bf16, row-major A, col-major B) ===============
// pa: lane-resolved A pointer  = A + (row0 + lane%16)*lda + (lane/16)*16
// pb: lane-resolved BT pointer = BT + (col0 + lane%16)*ldb + (lane/16)*16
__device__ __forceinline__ void mm_core(const bf16* __restrict__ pa, int lda,
                                        const bf16* __restrict__ pb, int ldb,
                                        int K, v8f acc[2][4]) {
#pragma unroll 1
  for (int k0 = 0; k0 < K; k0 += 32) {
    v16bf a0 = make_frag(*(const bf16x8*)(pa),            *(const bf16x8*)(pa + 8));
    v16bf a1 = make_frag(*(const bf16x8*)(pa + 16 * lda), *(const bf16x8*)(pa + 16 * lda + 8));
    v16bf b0 = *(const v16bf*)(pb);
    v16bf b1 = *(const v16bf*)(pb + 16 * ldb);
    v16bf b2 = *(const v16bf*)(pb + 32 * ldb);
    v16bf b3 = *(const v16bf*)(pb + 48 * ldb);
    acc[0][0] = WMMA_BF16(a0, b0, acc[0][0]);
    acc[0][1] = WMMA_BF16(a0, b1, acc[0][1]);
    acc[0][2] = WMMA_BF16(a0, b2, acc[0][2]);
    acc[0][3] = WMMA_BF16(a0, b3, acc[0][3]);
    acc[1][0] = WMMA_BF16(a1, b0, acc[1][0]);
    acc[1][1] = WMMA_BF16(a1, b1, acc[1][1]);
    acc[1][2] = WMMA_BF16(a1, b2, acc[1][2]);
    acc[1][3] = WMMA_BF16(a1, b3, acc[1][3]);
    pa += 32; pb += 32;
  }
}

// ================= Kernel 1: adjacency A = softmax(relu(E E^T)) ============
__global__ void __launch_bounds__(256, 1)
adj_kernel(const float* __restrict__ emb, bf16* __restrict__ A, bf16* __restrict__ AT) {
  __shared__ float embn[NE];
  __shared__ float red[256];
  int n = blockIdx.x;
  int t = threadIdx.x;
  if (t < NE) embn[t] = emb[n * NE + t];
  __syncthreads();

  float v[4];
  float lmax = -1e30f;
#pragma unroll
  for (int j = 0; j < 4; ++j) {
    int m = t + j * 256;
    float acc = 0.f;
#pragma unroll
    for (int d = 0; d < NE; ++d) acc += embn[d] * emb[m * NE + d];
    acc = fmaxf(acc, 0.f);
    v[j] = acc;
    lmax = fmaxf(lmax, acc);
  }
  red[t] = lmax; __syncthreads();
  for (int s = 128; s > 0; s >>= 1) { if (t < s) red[t] = fmaxf(red[t], red[t + s]); __syncthreads(); }
  float mx = red[0]; __syncthreads();

  float lsum = 0.f;
#pragma unroll
  for (int j = 0; j < 4; ++j) { v[j] = __expf(v[j] - mx); lsum += v[j]; }
  red[t] = lsum; __syncthreads();
  for (int s = 128; s > 0; s >>= 1) { if (t < s) red[t] += red[t + s]; __syncthreads(); }
  float inv = 1.f / red[0];

#pragma unroll
  for (int j = 0; j < 4; ++j) {
    int m = t + j * 256;
    bf16 a = (bf16)(v[j] * inv);
    A[(size_t)n * NN + m]  = a;
    AT[(size_t)m * NN + n] = a;
  }
}

// ========= Kernel 2: T = 2*A@A - I  (1024x1024x1024, bf16 WMMA) ============
__global__ void __launch_bounds__(256, 1)
cheb2_kernel(const bf16* __restrict__ A, const bf16* __restrict__ AT, bf16* __restrict__ T) {
  int lane = threadIdx.x & 31;
  int wave = threadIdx.x >> 5;                 // 8 waves: 4 row x 2 col
  int row0 = blockIdx.x * 128 + (wave >> 1) * 32;
  int col0 = blockIdx.y * 128 + (wave & 1) * 64;
  const bf16* pa = A  + (size_t)(row0 + (lane & 15)) * NN + ((lane >> 4) << 4);
  const bf16* pb = AT + (size_t)(col0 + (lane & 15)) * NN + ((lane >> 4) << 4);
  v8f acc[2][4] = {};
  mm_core(pa, NN, pb, NN, NN, acc);

  int rsub = (lane >> 4) << 3, c = lane & 15;
#pragma unroll
  for (int i = 0; i < 2; ++i)
#pragma unroll
    for (int j = 0; j < 4; ++j) {
      int col = col0 + 16 * j + c;
#pragma unroll
      for (int v = 0; v < 8; ++v) {
        int row = row0 + 16 * i + rsub + v;
        float val = 2.f * acc[i][j][v] - (row == col ? 1.f : 0.f);
        T[(size_t)row * NN + col] = (bf16)val;
      }
    }
}

// ===== Kernels 3/7: pack inputs -> XG[0][b][n][c] (bf16) and inpT[b][c][n] (bf16)
// block: 128 threads (one per channel c); grid: b * 128 groups of 8 nodes.
__global__ void __launch_bounds__(128, 1)
prep_gate_kernel(const float* __restrict__ x, const float* __restrict__ state,
                 bf16* __restrict__ XG0, bf16* __restrict__ inpT) {
  int bid = blockIdx.x;
  int b  = bid >> 7;
  int n0 = (bid & 127) * 8;
  int c  = threadIdx.x;
  bf16x8 tv;
#pragma unroll
  for (int q = 0; q < 8; ++q) {
    int n = n0 + q;
    float v = (c < 64) ? x[((size_t)b * NN + n) * 64 + c]
                       : state[((size_t)b * NN + n) * 64 + (c - 64)];
    bf16 h = (bf16)v;
    XG0[((size_t)b * NN + n) * NC + c] = h;      // coalesced across c
    tv[q] = h;
  }
  *(bf16x8*)(inpT + ((size_t)b * NC + c) * NN + n0) = tv;  // 16B vector store
}

__global__ void __launch_bounds__(128, 1)
prep_cand_kernel(const float* __restrict__ x, const float* __restrict__ state,
                 const float* __restrict__ zr,
                 bf16* __restrict__ XG0, bf16* __restrict__ inpT) {
  int bid = blockIdx.x;
  int b  = bid >> 7;
  int n0 = (bid & 127) * 8;
  int c  = threadIdx.x;
  bf16x8 tv;
#pragma unroll
  for (int q = 0; q < 8; ++q) {
    int n = n0 + q;
    float v;
    if (c < 64) v = x[((size_t)b * NN + n) * 64 + c];
    else        v = zr[((size_t)b * NN + n) * NC + c] *          // r gate (cols 64..127)
                    state[((size_t)b * NN + n) * 64 + (c - 64)];
    bf16 h = (bf16)v;
    XG0[((size_t)b * NN + n) * NC + c] = h;
    tv[q] = h;
  }
  *(bf16x8*)(inpT + ((size_t)b * NC + c) * NN + n0) = tv;
}

// ==== Kernels 4/8: XG[k][b] = S_k @ inp[b]  (S1=A, S2=T) =====================
__global__ void __launch_bounds__(256, 1)
support_kernel(const bf16* __restrict__ A, const bf16* __restrict__ T,
               const bf16* __restrict__ inpT, bf16* __restrict__ XG) {
  int b    = blockIdx.y;
  int kidx = blockIdx.z;                        // 0 -> A, 1 -> T
  const bf16* S  = (kidx == 0) ? A : T;
  const bf16* BT = inpT + (size_t)b * NC * NN;
  int lane = threadIdx.x & 31;
  int wave = threadIdx.x >> 5;
  int row0 = blockIdx.x * 128 + (wave >> 1) * 32;
  int col0 = (wave & 1) * 64;
  const bf16* pa = S  + (size_t)(row0 + (lane & 15)) * NN + ((lane >> 4) << 4);
  const bf16* pb = BT + (size_t)(col0 + (lane & 15)) * NN + ((lane >> 4) << 4);
  v8f acc[2][4] = {};
  mm_core(pa, NN, pb, NN, NN, acc);

  bf16* out = XG + (size_t)(kidx + 1) * XG_KSTRIDE + (size_t)b * XG_BSTRIDE;
  int rsub = (lane >> 4) << 3, c = lane & 15;
#pragma unroll
  for (int i = 0; i < 2; ++i)
#pragma unroll
    for (int j = 0; j < 4; ++j) {
      int col = col0 + 16 * j + c;
#pragma unroll
      for (int v = 0; v < 8; ++v) {
        int row = row0 + 16 * i + rsub + v;
        out[(size_t)row * NC + col] = (bf16)acc[i][j][v];
      }
    }
}

// == Kernels 5/9: per-node weights WT[n][o][ki] = sum_d emb[n,d] wp[d,k,i,o] ==
__global__ void __launch_bounds__(256, 1)
weights_kernel(const float* __restrict__ emb, const float* __restrict__ wp,
               int OUT, bf16* __restrict__ WT) {
  size_t idx = (size_t)blockIdx.x * 256 + threadIdx.x;
  size_t total = (size_t)NN * OUT * NKI;
  if (idx >= total) return;
  int ki = (int)(idx % NKI);
  int o  = (int)((idx / NKI) % OUT);
  int n  = (int)(idx / ((size_t)NKI * OUT));
  int k = ki >> 7, i = ki & 127;
  const float* w = wp + ((size_t)k * NC + i) * OUT + o;
  size_t dstride = (size_t)3 * NC * OUT;
  float acc = 0.f;
#pragma unroll
  for (int d = 0; d < NE; ++d) acc += emb[n * NE + d] * w[(size_t)d * dstride];
  WT[idx] = (bf16)acc;
}

// ======= Kernel 6: per-node gate GEMM (64x384 @ 384x128) + sigmoid =========
// block: 128 threads = 4 waves (2 row-halves x 2 col-halves)
__global__ void __launch_bounds__(128, 1)
gate_kernel(const bf16* __restrict__ XG, const bf16* __restrict__ WT,
            const float* __restrict__ emb, const float* __restrict__ bgate,
            float* __restrict__ zr) {
  int n = blockIdx.x;
  int lane = threadIdx.x & 31;
  int wave = threadIdx.x >> 5;
  int b0   = (wave >> 1) * 32;
  int col0 = (wave & 1) * 64;
  const bf16* xgn = XG + (size_t)n * NC;
  const bf16* pb  = WT + (size_t)n * NC * NKI + (size_t)(col0 + (lane & 15)) * NKI + ((lane >> 4) << 4);
  v8f acc[2][4] = {};
#pragma unroll 1
  for (int k = 0; k < 3; ++k) {
    const bf16* pa = xgn + (size_t)k * XG_KSTRIDE +
                     (size_t)(b0 + (lane & 15)) * XG_BSTRIDE + ((lane >> 4) << 4);
#pragma unroll 1
    for (int i0 = 0; i0 < NC; i0 += 32) {
      v16bf a0 = make_frag(*(const bf16x8*)(pa),
                           *(const bf16x8*)(pa + 8));
      v16bf a1 = make_frag(*(const bf16x8*)(pa + 16 * XG_BSTRIDE),
                           *(const bf16x8*)(pa + 16 * XG_BSTRIDE + 8));
      v16bf q0 = *(const v16bf*)(pb);
      v16bf q1 = *(const v16bf*)(pb + 16 * NKI);
      v16bf q2 = *(const v16bf*)(pb + 32 * NKI);
      v16bf q3 = *(const v16bf*)(pb + 48 * NKI);
      acc[0][0] = WMMA_BF16(a0, q0, acc[0][0]);
      acc[0][1] = WMMA_BF16(a0, q1, acc[0][1]);
      acc[0][2] = WMMA_BF16(a0, q2, acc[0][2]);
      acc[0][3] = WMMA_BF16(a0, q3, acc[0][3]);
      acc[1][0] = WMMA_BF16(a1, q0, acc[1][0]);
      acc[1][1] = WMMA_BF16(a1, q1, acc[1][1]);
      acc[1][2] = WMMA_BF16(a1, q2, acc[1][2]);
      acc[1][3] = WMMA_BF16(a1, q3, acc[1][3]);
      pa += 32; pb += 32;
    }
  }
  int rsub = (lane >> 4) << 3, c = lane & 15;
#pragma unroll
  for (int i = 0; i < 2; ++i)
#pragma unroll
    for (int j = 0; j < 4; ++j) {
      int o = col0 + 16 * j + c;
      float bias = 0.f;
#pragma unroll
      for (int d = 0; d < NE; ++d) bias += emb[n * NE + d] * bgate[d * NC + o];
#pragma unroll
      for (int v = 0; v < 8; ++v) {
        int b = b0 + 16 * i + rsub + v;
        float val = acc[i][j][v] + bias;
        val = 1.f / (1.f + __expf(-val));
        zr[((size_t)b * NN + n) * NC + o] = val;   // z: o<64, r: o>=64
      }
    }
}

// == Kernel 10: per-node update GEMM (64x384 @ 384x64) + tanh + GRU blend ===
// block: 64 threads = 2 row-waves
__global__ void __launch_bounds__(64, 1)
update_kernel(const bf16* __restrict__ XG, const bf16* __restrict__ WT,
              const float* __restrict__ emb, const float* __restrict__ bupd,
              const float* __restrict__ zr, const float* __restrict__ state,
              float* __restrict__ out) {
  int n = blockIdx.x;
  int lane = threadIdx.x & 31;
  int wave = threadIdx.x >> 5;
  int b0   = wave * 32;
  const bf16* xgn = XG + (size_t)n * NC;
  const bf16* pb  = WT + (size_t)n * 64 * NKI + (size_t)(lane & 15) * NKI + ((lane >> 4) << 4);
  v8f acc[2][4] = {};
#pragma unroll 1
  for (int k = 0; k < 3; ++k) {
    const bf16* pa = xgn + (size_t)k * XG_KSTRIDE +
                     (size_t)(b0 + (lane & 15)) * XG_BSTRIDE + ((lane >> 4) << 4);
#pragma unroll 1
    for (int i0 = 0; i0 < NC; i0 += 32) {
      v16bf a0 = make_frag(*(const bf16x8*)(pa),
                           *(const bf16x8*)(pa + 8));
      v16bf a1 = make_frag(*(const bf16x8*)(pa + 16 * XG_BSTRIDE),
                           *(const bf16x8*)(pa + 16 * XG_BSTRIDE + 8));
      v16bf q0 = *(const v16bf*)(pb);
      v16bf q1 = *(const v16bf*)(pb + 16 * NKI);
      v16bf q2 = *(const v16bf*)(pb + 32 * NKI);
      v16bf q3 = *(const v16bf*)(pb + 48 * NKI);
      acc[0][0] = WMMA_BF16(a0, q0, acc[0][0]);
      acc[0][1] = WMMA_BF16(a0, q1, acc[0][1]);
      acc[0][2] = WMMA_BF16(a0, q2, acc[0][2]);
      acc[0][3] = WMMA_BF16(a0, q3, acc[0][3]);
      acc[1][0] = WMMA_BF16(a1, q0, acc[1][0]);
      acc[1][1] = WMMA_BF16(a1, q1, acc[1][1]);
      acc[1][2] = WMMA_BF16(a1, q2, acc[1][2]);
      acc[1][3] = WMMA_BF16(a1, q3, acc[1][3]);
      pa += 32; pb += 32;
    }
  }
  int rsub = (lane >> 4) << 3, c = lane & 15;
#pragma unroll
  for (int i = 0; i < 2; ++i)
#pragma unroll
    for (int j = 0; j < 4; ++j) {
      int o = 16 * j + c;
      float bias = 0.f;
#pragma unroll
      for (int d = 0; d < NE; ++d) bias += emb[n * NE + d] * bupd[d * 64 + o];
#pragma unroll
      for (int v = 0; v < 8; ++v) {
        int b = b0 + 16 * i + rsub + v;
        float hh = tanhf(acc[i][j][v] + bias);
        float z  = zr[((size_t)b * NN + n) * NC + o];
        float s  = state[((size_t)b * NN + n) * 64 + o];
        out[((size_t)b * NN + n) * 64 + o] = z * s + (1.f - z) * hh;
      }
    }
}

// ============================== launch =====================================
extern "C" void kernel_launch(void* const* d_in, const int* in_sizes, int n_in,
                              void* d_out, int out_size, void* d_ws, size_t ws_size,
                              hipStream_t stream) {
  const float* x     = (const float*)d_in[0];  // (64,1024,64)
  const float* state = (const float*)d_in[1];  // (64,1024,64)
  const float* emb   = (const float*)d_in[2];  // (1024,16)
  const float* Wg    = (const float*)d_in[3];  // (16,3,128,128)
  const float* bg    = (const float*)d_in[4];  // (16,128)
  const float* Wu    = (const float*)d_in[5];  // (16,3,128,64)
  const float* bu    = (const float*)d_in[6];  // (16,64)
  float* out = (float*)d_out;                  // (64,1024,64)

  char* ws = (char*)d_ws;
  const size_t MB = 1024 * 1024;
  bf16*  A    = (bf16*)(ws + 0 * MB);    //   2 MB adjacency, row-major
  bf16*  AT   = (bf16*)(ws + 2 * MB);    //   2 MB adjacency, col-major
  bf16*  T    = (bf16*)(ws + 4 * MB);    //   2 MB 2A@A - I
  bf16*  inpT = (bf16*)(ws + 6 * MB);    //  16 MB inp[b] col-major
  bf16*  XG   = (bf16*)(ws + 22 * MB);   //  48 MB XG[k][b][n][c]
  bf16*  WT   = (bf16*)(ws + 70 * MB);   //  96 MB per-node weights (reused)
  float* zr   = (float*)(ws + 166 * MB); //  32 MB sigmoid(z|r)
  (void)ws_size; (void)in_sizes; (void)n_in; (void)out_size; // 198 MB total

  // ---- shared graph structure ----
  adj_kernel  <<<NN, 256, 0, stream>>>(emb, A, AT);
  cheb2_kernel<<<dim3(8, 8), 256, 0, stream>>>(A, AT, T);

  // ---- gate GCN: z_r = sigmoid(cheb_gcn(concat(x,state))) ----
  prep_gate_kernel<<<NB * 128, 128, 0, stream>>>(x, state, XG, inpT);
  support_kernel  <<<dim3(8, NB, 2), 256, 0, stream>>>(A, T, inpT, XG);
  {
    size_t total = (size_t)NN * NC * NKI;
    weights_kernel<<<(unsigned)((total + 255) / 256), 256, 0, stream>>>(emb, Wg, NC, WT);
  }
  gate_kernel<<<NN, 128, 0, stream>>>(XG, WT, emb, bg, zr);

  // ---- update GCN: h_hat = tanh(cheb_gcn(concat(x, r*state))) ----
  prep_cand_kernel<<<NB * 128, 128, 0, stream>>>(x, state, zr, XG, inpT);
  support_kernel  <<<dim3(8, NB, 2), 256, 0, stream>>>(A, T, inpT, XG);
  {
    size_t total = (size_t)NN * 64 * NKI;
    weights_kernel<<<(unsigned)((total + 255) / 256), 256, 0, stream>>>(emb, Wu, 64, WT);
  }
  update_kernel<<<NN, 64, 0, stream>>>(XG, WT, emb, bu, zr, state, out);
}
